// CausalSelfAttention_34411277976199
// MI455X (gfx1250) — matrix operand order repaired and checked by reading, not compiled, not run on hardware
//
#include <hip/hip_runtime.h>

// Causal self-attention forward for MI455X (gfx1250, wave32, WMMA).
// B=4, S=2048, D=1024, H=16, HS=64.
// Pipeline: f32->bf16 converts, QKV GEMM (bf16 WMMA, f32 acc, LDS-staged B,
// double buffered), flash-style causal attention (block-shared LDS K/V tiles,
// online softmax), proj GEMM -> f32.
// Workspace requirement: ~92.3 MB.

#define BB 4
#define SS 2048
#define DD 1024
#define HH 16
#define HSZ 64

typedef __attribute__((ext_vector_type(16))) __bf16        v16bf;
typedef __attribute__((ext_vector_type(8)))  float         v8f;
typedef __attribute__((ext_vector_type(8)))  unsigned int  v8u;
typedef unsigned short ushort_t;

__device__ __forceinline__ ushort_t f2bf(float f) {
  unsigned int u = __builtin_bit_cast(unsigned int, f);
  u += 0x7FFFu + ((u >> 16) & 1u);          // round-to-nearest-even
  return (ushort_t)(u >> 16);
}

__device__ __forceinline__ v8f wmma_bf16(v16bf a, v16bf b, v8f c) {
  // (neg_a, A, neg_b, B, c_mod, C, reuse_a, reuse_b)
  return __builtin_amdgcn_wmma_f32_16x16x32_bf16(false, a, false, b,
                                                 (short)0, c, false, false);
}

// A-matrix fragment, 16x32 bf16. rowptr -> this lane's row (row = lane&15)
// at k0. Lane dword i holds K = kbase, kbase+1:
//   kbase = 2i + 16*hi   (one contiguous 32B run)
__device__ __forceinline__ v16bf load_a_frag(const ushort_t* rowptr, int hi) {
  v8u f;
#pragma unroll
  for (int i = 0; i < 8; ++i) {
    int kb = 2 * i + 16 * hi;
    f[i] = *(const unsigned int*)(rowptr + kb);
  }
  return __builtin_bit_cast(v16bf, f);
}

// B-matrix fragment, 32x16 bf16, from K-contiguous storage (column n's K-run
// contiguous). Column = lane&15; dword i holds K = 2i + 16*hi, +1.
__device__ __forceinline__ v16bf load_b_frag(const ushort_t* colptr, int hi) {
  v8u f;
#pragma unroll
  for (int i = 0; i < 8; ++i) {
    f[i] = *(const unsigned int*)(colptr + 2 * i + 16 * hi);
  }
  return __builtin_bit_cast(v16bf, f);
}

// ---------------------------------------------------------------- converts
__global__ void k_cvt_bf16(const float* __restrict__ in,
                           ushort_t* __restrict__ out, int n) {
  int i = blockIdx.x * blockDim.x + threadIdx.x;
  if (i < n) out[i] = f2bf(in[i]);
}

// in: [K,N] row-major f32  ->  out: [N,K] bf16 (each column's K-run contiguous)
__global__ void k_cvt_t_bf16(const float* __restrict__ in,
                             ushort_t* __restrict__ out, int K, int N) {
  long idx = (long)blockIdx.x * blockDim.x + threadIdx.x;
  long total = (long)K * (long)N;
  if (idx < total) {
    int k = (int)(idx % K);
    int n = (int)(idx / K);
    out[idx] = f2bf(in[(long)k * N + n]);
  }
}

// ---------------------------------------------------------------- QKV GEMM
// Block: 256 rows x 64 cols, 8 waves; wave: 32 rows x 64 cols (8 WMMA/K-step).
// B tile staged in LDS (double buffered), shared by all waves in the block.
__global__ void __launch_bounds__(256) k_qkv_gemm(
    const ushort_t* __restrict__ X,    // [B*S, D] bf16 row-major
    const ushort_t* __restrict__ WT,   // [3D, D] bf16, column-K contiguous
    const float* __restrict__ bias,    // [3D]
    ushort_t* __restrict__ Qb, ushort_t* __restrict__ Kb,
    ushort_t* __restrict__ Vb) {
  __shared__ ushort_t bs[2][64][40];   // 64 cols x 32 k (+8 pad)
  const int tid  = threadIdx.x;
  const int lane = tid & 31;
  const int wv   = tid >> 5;
  const int hi   = (lane >> 4) & 1;
  const int l16  = lane & 15;
  const int m0   = blockIdx.y * 256 + wv * 32;
  const int n0   = blockIdx.x * 64;
  const int lcol = tid >> 2;           // 0..63
  const int lchk = tid & 3;            // 16B chunk within a column's K-run

  v8f zero = {0.f, 0.f, 0.f, 0.f, 0.f, 0.f, 0.f, 0.f};
  v8f acc[2][4];
#pragma unroll
  for (int ms = 0; ms < 2; ++ms)
#pragma unroll
    for (int t = 0; t < 4; ++t) acc[ms][t] = zero;

  const ushort_t* arow0 = X + (size_t)(m0 + l16) * DD;
  const ushort_t* arow1 = X + (size_t)(m0 + 16 + l16) * DD;
  const ushort_t* bsrc  = WT + (size_t)(n0 + lcol) * DD + lchk * 8;

  *(uint4*)(&bs[0][lcol][lchk * 8]) = *(const uint4*)(bsrc);

  for (int kk = 0, step = 0; kk < DD; kk += 32, ++step) {
    const int cur = step & 1;
    __syncthreads();
    if (kk + 32 < DD) {
      *(uint4*)(&bs[cur ^ 1][lcol][lchk * 8]) =
          *(const uint4*)(bsrc + kk + 32);
    }
    if (kk + 64 < DD) __builtin_prefetch(arow0 + kk + 64, 0, 1);
    v16bf a0 = load_a_frag(arow0 + kk, hi);
    v16bf a1 = load_a_frag(arow1 + kk, hi);
#pragma unroll
    for (int t = 0; t < 4; ++t) {
      v16bf b = load_b_frag(&bs[cur][t * 16 + l16][0], hi);
      acc[0][t] = wmma_bf16(a0, b, acc[0][t]);
      acc[1][t] = wmma_bf16(a1, b, acc[1][t]);
    }
  }

#pragma unroll
  for (int ms = 0; ms < 2; ++ms) {
#pragma unroll
    for (int t = 0; t < 4; ++t) {
      int col   = n0 + 16 * t + l16;
      float bv  = bias[col];
      int which = col >> 10;       // 0=Q 1=K 2=V
      int d     = col & 1023;
      int h     = d >> 6, hs = d & 63;
#pragma unroll
      for (int r = 0; r < 8; ++r) {
        int row = m0 + ms * 16 + r + 8 * hi;
        int bi  = row >> 11;       // / 2048
        int s   = row & 2047;
        ushort_t val = f2bf(acc[ms][t][r] + bv);
        if (which == 0)
          Qb[(((size_t)bi * HH + h) * SS + s) * HSZ + hs] = val;
        else if (which == 1)
          Kb[(((size_t)bi * HH + h) * SS + s) * HSZ + hs] = val;
        else
          Vb[(((size_t)bi * HH + h) * HSZ + hs) * SS + s] = val;
      }
    }
  }
}

// ---------------------------------------------------------------- attention
// Block = 8 waves = 8 consecutive 16-query tiles of one (b,h). K/V 32-key
// tiles staged in shared LDS (double buffered) and consumed by all waves;
// loop trip count uniform per block, per-wave compute causally predicated.
__global__ void __launch_bounds__(256) k_attn(
    const ushort_t* __restrict__ Qb,   // [B,H,S,HS]
    const ushort_t* __restrict__ Kb,   // [B,H,S,HS]
    const ushort_t* __restrict__ Vb,   // [B,H,HS,S]
    ushort_t* __restrict__ Ob) {       // [B*S, D] bf16
  __shared__ ushort_t ks[2][32][72];   // 32 keys x 64 hs (+8 pad)
  __shared__ ushort_t vs[2][64][40];   // 64 hs x 32 keys (+8 pad)
  __shared__ ushort_t plds[8][512];    // per-wave P re-layout buffer
  const int tid  = threadIdx.x;
  const int lane = tid & 31;
  const int wv   = tid >> 5;
  const int hi   = (lane >> 4) & 1;
  const int l16  = lane & 15;
  const int b    = blockIdx.z, h = blockIdx.y;
  const int q0   = (blockIdx.x * 8 + wv) * 16;
  const int qend = (blockIdx.x * 8 + 7) * 16 + 15;  // block's last query

  const ushort_t* Qh = Qb + (size_t)(b * HH + h) * SS * HSZ;
  const ushort_t* Kh = Kb + (size_t)(b * HH + h) * SS * HSZ;
  const ushort_t* Vh = Vb + (size_t)(b * HH + h) * HSZ * SS;

  // cooperative staging indices
  const int kkey = tid >> 3, kchk = tid & 7;   // K: 32 rows x 8 chunks
  const int vhs  = tid >> 2, vchk = tid & 3;   // V: 64 rows x 4 chunks

  v16bf aq0 = load_a_frag(Qh + (size_t)(q0 + l16) * HSZ, hi);
  v16bf aq1 = load_a_frag(Qh + (size_t)(q0 + l16) * HSZ + 32, hi);

  v8f zero = {0.f, 0.f, 0.f, 0.f, 0.f, 0.f, 0.f, 0.f};
  v8f accO[4];
#pragma unroll
  for (int t = 0; t < 4; ++t) accO[t] = zero;
  float mrow[8], lrow[8];
#pragma unroll
  for (int r = 0; r < 8; ++r) { mrow[r] = -3.0e38f; lrow[r] = 0.f; }

  ushort_t* pl = &plds[wv][0];
  const float scale = 0.25f;  // reference scales by 1/sqrt(H), H=16

  // preload kv step 0
  *(uint4*)(&ks[0][kkey][kchk * 8]) =
      *(const uint4*)(Kh + (size_t)kkey * HSZ + kchk * 8);
  *(uint4*)(&vs[0][vhs][vchk * 8]) =
      *(const uint4*)(Vh + (size_t)vhs * SS + vchk * 8);

  for (int n0 = 0, step = 0; n0 <= qend; n0 += 32, ++step) {
    const int cur = step & 1;
    __syncthreads();
    if (n0 + 32 <= qend) {
      const int nn = n0 + 32;
      *(uint4*)(&ks[cur ^ 1][kkey][kchk * 8]) =
          *(const uint4*)(Kh + (size_t)(nn + kkey) * HSZ + kchk * 8);
      *(uint4*)(&vs[cur ^ 1][vhs][vchk * 8]) =
          *(const uint4*)(Vh + (size_t)vhs * SS + nn + vchk * 8);
    }
    if (n0 <= q0 + 15) {
      float p0[8], p1[8];
      {  // S tile 0: keys n0 .. n0+15
        v8f c = zero;
        c = wmma_bf16(aq0, load_b_frag(&ks[cur][l16][0], hi), c);
        c = wmma_bf16(aq1, load_b_frag(&ks[cur][l16][32], hi), c);
        int key = n0 + l16;
#pragma unroll
        for (int r = 0; r < 8; ++r) {
          int qrow = q0 + r + 8 * hi;
          p0[r] = (key <= qrow) ? c[r] * scale : -3.0e38f;
        }
      }
      {  // S tile 1: keys n0+16 .. n0+31
        v8f c = zero;
        c = wmma_bf16(aq0, load_b_frag(&ks[cur][16 + l16][0], hi), c);
        c = wmma_bf16(aq1, load_b_frag(&ks[cur][16 + l16][32], hi), c);
        int key = n0 + 16 + l16;
#pragma unroll
        for (int r = 0; r < 8; ++r) {
          int qrow = q0 + r + 8 * hi;
          p1[r] = (key <= qrow) ? c[r] * scale : -3.0e38f;
        }
      }
      // online softmax update (row M = r + 8*hi spans lanes of a half-group)
#pragma unroll
      for (int r = 0; r < 8; ++r) {
        float v = fmaxf(p0[r], p1[r]);
#pragma unroll
        for (int o = 1; o < 16; o <<= 1) v = fmaxf(v, __shfl_xor(v, o, 32));
        float mn    = fmaxf(mrow[r], v);
        float alpha = __expf(mrow[r] - mn);
        mrow[r] = mn;
        p0[r] = __expf(p0[r] - mn);
        p1[r] = __expf(p1[r] - mn);
        float rs = p0[r] + p1[r];
#pragma unroll
        for (int o = 1; o < 16; o <<= 1) rs += __shfl_xor(rs, o, 32);
        lrow[r] = lrow[r] * alpha + rs;
#pragma unroll
        for (int t = 0; t < 4; ++t) accO[t][r] = accO[t][r] * alpha;
      }
      // P: C-layout -> A-layout via per-wave LDS round-trip
#pragma unroll
      for (int r = 0; r < 8; ++r) {
        int M = r + 8 * hi;
        pl[M * 32 + l16]      = f2bf(p0[r]);
        pl[M * 32 + 16 + l16] = f2bf(p1[r]);
      }
      __builtin_amdgcn_wave_barrier();
      v16bf ap = load_a_frag(pl + l16 * 32, hi);
      __builtin_amdgcn_wave_barrier();
      // accO += P(16x32) @ V(32x64)
#pragma unroll
      for (int t = 0; t < 4; ++t) {
        accO[t] = wmma_bf16(ap, load_b_frag(&vs[cur][t * 16 + l16][0], hi),
                            accO[t]);
      }
    }
  }

#pragma unroll
  for (int t = 0; t < 4; ++t) {
#pragma unroll
    for (int r = 0; r < 8; ++r) {
      int row   = q0 + r + 8 * hi;
      float inv = 1.0f / lrow[r];
      Ob[((size_t)b * SS + row) * DD + h * HSZ + t * 16 + l16] =
          f2bf(accO[t][r] * inv);
    }
  }
}

// ---------------------------------------------------------------- proj GEMM
__global__ void __launch_bounds__(256) k_proj_gemm(
    const ushort_t* __restrict__ Xb,   // [B*S, D] bf16 (attention output)
    const ushort_t* __restrict__ WT,   // [D, D] column-K contiguous
    const float* __restrict__ bias,
    float* __restrict__ out) {
  __shared__ ushort_t bs[2][64][40];
  const int tid  = threadIdx.x;
  const int lane = tid & 31;
  const int wv   = tid >> 5;
  const int hi   = (lane >> 4) & 1;
  const int l16  = lane & 15;
  const int m0   = blockIdx.y * 256 + wv * 32;
  const int n0   = blockIdx.x * 64;
  const int lcol = tid >> 2;
  const int lchk = tid & 3;

  v8f zero = {0.f, 0.f, 0.f, 0.f, 0.f, 0.f, 0.f, 0.f};
  v8f acc[2][4];
#pragma unroll
  for (int ms = 0; ms < 2; ++ms)
#pragma unroll
    for (int t = 0; t < 4; ++t) acc[ms][t] = zero;

  const ushort_t* arow0 = Xb + (size_t)(m0 + l16) * DD;
  const ushort_t* arow1 = Xb + (size_t)(m0 + 16 + l16) * DD;
  const ushort_t* bsrc  = WT + (size_t)(n0 + lcol) * DD + lchk * 8;

  *(uint4*)(&bs[0][lcol][lchk * 8]) = *(const uint4*)(bsrc);

  for (int kk = 0, step = 0; kk < DD; kk += 32, ++step) {
    const int cur = step & 1;
    __syncthreads();
    if (kk + 32 < DD) {
      *(uint4*)(&bs[cur ^ 1][lcol][lchk * 8]) =
          *(const uint4*)(bsrc + kk + 32);
    }
    if (kk + 64 < DD) __builtin_prefetch(arow0 + kk + 64, 0, 1);
    v16bf a0 = load_a_frag(arow0 + kk, hi);
    v16bf a1 = load_a_frag(arow1 + kk, hi);
#pragma unroll
    for (int t = 0; t < 4; ++t) {
      v16bf b = load_b_frag(&bs[cur][t * 16 + l16][0], hi);
      acc[0][t] = wmma_bf16(a0, b, acc[0][t]);
      acc[1][t] = wmma_bf16(a1, b, acc[1][t]);
    }
  }
#pragma unroll
  for (int ms = 0; ms < 2; ++ms) {
#pragma unroll
    for (int t = 0; t < 4; ++t) {
      float bv = bias[n0 + 16 * t + l16];
#pragma unroll
      for (int r = 0; r < 8; ++r) {
        out[(size_t)(m0 + ms * 16 + r + 8 * hi) * DD + n0 + 16 * t + l16] =
            acc[ms][t][r] + bv;
      }
    }
  }
}

// ---------------------------------------------------------------- launcher
extern "C" void kernel_launch(void* const* d_in, const int* in_sizes, int n_in,
                              void* d_out, int out_size, void* d_ws,
                              size_t ws_size, hipStream_t stream) {
  const float* x      = (const float*)d_in[0];
  const float* w_qkv  = (const float*)d_in[1];
  const float* b_qkv  = (const float*)d_in[2];
  const float* w_proj = (const float*)d_in[3];
  const float* b_proj = (const float*)d_in[4];
  float* out          = (float*)d_out;

  char* ws = (char*)d_ws;
  ushort_t* Xbf    = (ushort_t*)ws; ws += (size_t)BB * SS * DD * 2;
  ushort_t* WqkvT  = (ushort_t*)ws; ws += (size_t)3 * DD * DD * 2;
  ushort_t* WprojT = (ushort_t*)ws; ws += (size_t)DD * DD * 2;
  ushort_t* Qb     = (ushort_t*)ws; ws += (size_t)BB * HH * SS * HSZ * 2;
  ushort_t* Kb     = (ushort_t*)ws; ws += (size_t)BB * HH * SS * HSZ * 2;
  ushort_t* Vb     = (ushort_t*)ws; ws += (size_t)BB * HH * SS * HSZ * 2;
  ushort_t* Obf    = (ushort_t*)ws; ws += (size_t)BB * SS * DD * 2;

  int nX = BB * SS * DD;
  k_cvt_bf16<<<(nX + 255) / 256, 256, 0, stream>>>(x, Xbf, nX);
  long nWq = (long)DD * 3 * DD;
  k_cvt_t_bf16<<<(int)((nWq + 255) / 256), 256, 0, stream>>>(w_qkv, WqkvT, DD,
                                                             3 * DD);
  long nWp = (long)DD * DD;
  k_cvt_t_bf16<<<(int)((nWp + 255) / 256), 256, 0, stream>>>(w_proj, WprojT,
                                                             DD, DD);
  k_qkv_gemm<<<dim3(48, 32), 256, 0, stream>>>(Xbf, WqkvT, b_qkv, Qb, Kb, Vb);
  k_attn<<<dim3(SS / 16 / 8, HH, BB), 256, 0, stream>>>(Qb, Kb, Vb, Obf);
  k_proj_gemm<<<dim3(16, 32), 256, 0, stream>>>(Obf, WprojT, b_proj, out);
}
